// ClusterMemory_30279519437290
// MI455X (gfx1250) — compile-verified, matrix operand order checked by reading, not run on hardware
//
#include <hip/hip_runtime.h>

// ---------------------------------------------------------------------------
// ClusterMemory forward for MI455X (gfx1250, wave32, WMMA):
//   out[B,N] = (l2norm(inputs[B,D]) @ features[N,D]^T) * (1/TEMP)
//
// Pass 1: row-normalize inputs, quantize to bf16 into workspace (4 MB).
// Pass 2: one-shot convert features fp32->bf16 into workspace (64 MB).
// Pass 3: 128x128-tile WMMA bf16 GEMM, fp32 accumulate, x20 scale on store.
//         Staging is kept in named scalar registers (no arrays/lambdas) so
//         promote-alloca cannot demote the prefetch to LDS.
// ---------------------------------------------------------------------------

typedef __attribute__((ext_vector_type(16))) __bf16 v16bf;
typedef __attribute__((ext_vector_type(8)))  __bf16 v8bf;
typedef __attribute__((ext_vector_type(8)))  float  v8f;

#define TILE_M 128
#define TILE_N 128
#define TILE_K 32

static constexpr float INV_TEMP = 20.0f;   // 1 / 0.05

// round-to-nearest-even f32 -> bf16, two at a time, pure 32-bit ALU ops
__device__ __forceinline__ unsigned pack2_bf16(float a, float b) {
  unsigned ua = __builtin_bit_cast(unsigned, a);
  unsigned ub = __builtin_bit_cast(unsigned, b);
  ua += 0x7FFFu + ((ua >> 16) & 1u);
  ub += 0x7FFFu + ((ub >> 16) & 1u);
  return (ua >> 16) | (ub & 0xFFFF0000u);
}

// -------------------------- Pass 1: normalize ------------------------------
__global__ __launch_bounds__(256) void normalize_rows_bf16(
    const float* __restrict__ x, unsigned short* __restrict__ xbf, int D) {
  const int row = blockIdx.x;
  const float4* xv = reinterpret_cast<const float4*>(x + (size_t)row * D);
  const int nvec = D >> 2;

  float s = 0.0f;
  for (int i = threadIdx.x; i < nvec; i += blockDim.x) {
    float4 v = xv[i];
    s += v.x * v.x + v.y * v.y + v.z * v.z + v.w * v.w;
  }
  for (int off = 16; off > 0; off >>= 1) s += __shfl_xor(s, off, 32);

  __shared__ float warpsum[8];
  const int wave = threadIdx.x >> 5;
  if ((threadIdx.x & 31) == 0) warpsum[wave] = s;
  __syncthreads();
  float tot = 0.0f;
#pragma unroll
  for (int w = 0; w < 8; ++w) tot += warpsum[w];

  const float inv = 1.0f / fmaxf(sqrtf(tot), 1e-12f);

  for (int i = threadIdx.x; i < nvec; i += blockDim.x) {
    float4 v = xv[i];
    uint2 pk;
    pk.x = pack2_bf16(v.x * inv, v.y * inv);
    pk.y = pack2_bf16(v.z * inv, v.w * inv);
    *reinterpret_cast<uint2*>(xbf + (size_t)row * D + (size_t)i * 4) = pk;
  }
}

// --------------------- Pass 2: bulk fp32 -> bf16 ---------------------------
__global__ __launch_bounds__(256) void convert_f32_to_bf16(
    const float* __restrict__ src, unsigned short* __restrict__ dst,
    size_t nvec4) {
  const float4* sv = reinterpret_cast<const float4*>(src);
  size_t i = (size_t)blockIdx.x * blockDim.x + threadIdx.x;
  const size_t stride = (size_t)gridDim.x * blockDim.x;
  for (; i < nvec4; i += stride) {
    float4 v = sv[i];
    uint2 pk;
    pk.x = pack2_bf16(v.x, v.y);
    pk.y = pack2_bf16(v.z, v.w);
    *reinterpret_cast<uint2*>(dst + i * 4) = pk;
  }
}

// -------------------------- Pass 3: WMMA GEMM ------------------------------
template <bool B_IS_BF16>
__global__ __launch_bounds__(256) void cosine_gemm_wmma(
    const unsigned short* __restrict__ Abf,   // [B][D] bf16 normalized inputs
    const unsigned short* __restrict__ Bbf,   // [N][D] bf16 bank (fast path)
    const float* __restrict__ Bf32,           // [N][D] fp32 bank (fallback)
    float* __restrict__ Out,                  // [B][N] fp32
    int D, int N) {
  __shared__ unsigned short ldsA[TILE_M * TILE_K];   // 8 KB
  __shared__ unsigned short ldsB[TILE_N * TILE_K];   // 8 KB

  const int t      = threadIdx.x;
  const int lane   = t & 31;
  const int wave   = t >> 5;
  const int wave_m = wave >> 2;   // 0..1 -> 64-row strip
  const int wave_n = wave & 3;    // 0..3 -> 32-col strip

  const size_t rowA0 = (size_t)blockIdx.y * TILE_M;
  const size_t rowB0 = (size_t)blockIdx.x * TILE_N;

  // global staging: 4 threads per 32-element row, 8 elements each, 2 passes
  const int lr = t >> 2;          // 0..63
  const int lc = (t & 3) << 3;    // 0,8,16,24

  // per-thread global source pointers, advanced by TILE_K each iteration
  const unsigned short* pA0 = Abf + (rowA0 + (size_t)lr) * D + lc;
  const unsigned short* pA1 = pA0 + (size_t)64 * D;
  const unsigned short* pBb0 = nullptr;
  const unsigned short* pBb1 = nullptr;
  const float*          pBf0 = nullptr;
  const float*          pBf1 = nullptr;
  if constexpr (B_IS_BF16) {
    pBb0 = Bbf + (rowB0 + (size_t)lr) * D + lc;
    pBb1 = pBb0 + (size_t)64 * D;
  } else {
    pBf0 = Bf32 + (rowB0 + (size_t)lr) * D + lc;
    pBf1 = pBf0 + (size_t)64 * D;
  }

  // per-thread LDS destination pointers (constant across iterations)
  unsigned short* stA0 = &ldsA[lr * TILE_K + lc];
  unsigned short* stA1 = stA0 + 64 * TILE_K;
  unsigned short* stB0 = &ldsB[lr * TILE_K + lc];
  unsigned short* stB1 = stB0 + 64 * TILE_K;

  // staged tile data: individually named scalars -> guaranteed VGPRs
  uint4  sa0, sa1;
  uint4  sb0, sb1;
  float4 f00, f01, f10, f11;

  // prologue: load first K-tile
  sa0 = *reinterpret_cast<const uint4*>(pA0);
  sa1 = *reinterpret_cast<const uint4*>(pA1);
  pA0 += TILE_K; pA1 += TILE_K;
  if constexpr (B_IS_BF16) {
    sb0 = *reinterpret_cast<const uint4*>(pBb0);
    sb1 = *reinterpret_cast<const uint4*>(pBb1);
    pBb0 += TILE_K; pBb1 += TILE_K;
  } else {
    f00 = *reinterpret_cast<const float4*>(pBf0);
    f01 = *reinterpret_cast<const float4*>(pBf0 + 4);
    f10 = *reinterpret_cast<const float4*>(pBf1);
    f11 = *reinterpret_cast<const float4*>(pBf1 + 4);
    pBf0 += TILE_K; pBf1 += TILE_K;
  }

  // ISA 16-bit fragment layout: lanes 0-15 take K[0..7]+K[16..23] of row
  // (lane&15); lanes 16-31 take K[8..15]+K[24..31].
  const int fr   = lane & 15;
  const int kofs = (lane >> 4) << 3;   // 0 or 8 elements
  const unsigned short* frA = &ldsA[(wave_m * 64 + fr) * TILE_K + kofs];
  const unsigned short* frB = &ldsB[(wave_n * 32 + fr) * TILE_K + kofs];

  v8f acc[4][2] = {};
  const int ktiles = D / TILE_K;

  for (int kt = 0; kt < ktiles; ++kt) {
    // commit staged tile to LDS (waits for outstanding global loads here)
    *reinterpret_cast<uint4*>(stA0) = sa0;
    *reinterpret_cast<uint4*>(stA1) = sa1;
    if constexpr (B_IS_BF16) {
      *reinterpret_cast<uint4*>(stB0) = sb0;
      *reinterpret_cast<uint4*>(stB1) = sb1;
    } else {
      uint4 p0, p1;
      p0.x = pack2_bf16(f00.x, f00.y);
      p0.y = pack2_bf16(f00.z, f00.w);
      p0.z = pack2_bf16(f01.x, f01.y);
      p0.w = pack2_bf16(f01.z, f01.w);
      p1.x = pack2_bf16(f10.x, f10.y);
      p1.y = pack2_bf16(f10.z, f10.w);
      p1.z = pack2_bf16(f11.x, f11.y);
      p1.w = pack2_bf16(f11.z, f11.w);
      *reinterpret_cast<uint4*>(stB0) = p0;
      *reinterpret_cast<uint4*>(stB1) = p1;
    }
    __syncthreads();

    // issue next tile's global loads; no wait until next store_lds
    if (kt + 1 < ktiles) {
      sa0 = *reinterpret_cast<const uint4*>(pA0);
      sa1 = *reinterpret_cast<const uint4*>(pA1);
      pA0 += TILE_K; pA1 += TILE_K;
      if constexpr (B_IS_BF16) {
        sb0 = *reinterpret_cast<const uint4*>(pBb0);
        sb1 = *reinterpret_cast<const uint4*>(pBb1);
        pBb0 += TILE_K; pBb1 += TILE_K;
      } else {
        f00 = *reinterpret_cast<const float4*>(pBf0);
        f01 = *reinterpret_cast<const float4*>(pBf0 + 4);
        f10 = *reinterpret_cast<const float4*>(pBf1);
        f11 = *reinterpret_cast<const float4*>(pBf1 + 4);
        pBf0 += TILE_K; pBf1 += TILE_K;
      }
    }

    // fragments + 8 WMMAs
    v16bf af[4];
#pragma unroll
    for (int i = 0; i < 4; ++i) {
      const unsigned short* rp = frA + i * 16 * TILE_K;
      v8bf lo = *reinterpret_cast<const v8bf*>(rp);
      v8bf hi = *reinterpret_cast<const v8bf*>(rp + 16);
      af[i] = __builtin_shufflevector(lo, hi, 0, 1, 2, 3, 4, 5, 6, 7,
                                      8, 9, 10, 11, 12, 13, 14, 15);
    }
    v16bf bfm[2];
#pragma unroll
    for (int j = 0; j < 2; ++j) {
      const unsigned short* rp = frB + j * 16 * TILE_K;
      v8bf lo = *reinterpret_cast<const v8bf*>(rp);
      v8bf hi = *reinterpret_cast<const v8bf*>(rp + 16);
      bfm[j] = __builtin_shufflevector(lo, hi, 0, 1, 2, 3, 4, 5, 6, 7,
                                       8, 9, 10, 11, 12, 13, 14, 15);
    }

#pragma unroll
    for (int i = 0; i < 4; ++i)
#pragma unroll
      for (int j = 0; j < 2; ++j)
        acc[i][j] = __builtin_amdgcn_wmma_f32_16x16x32_bf16(
            false, af[i], false, bfm[j], (short)0, acc[i][j], false, false);

    __syncthreads();
  }

  // C/D layout: VGPR e -> row e + 8*(lane>>4), col (lane&15)
  const int orow = (lane >> 4) * 8;
  const int ocol = lane & 15;
#pragma unroll
  for (int i = 0; i < 4; ++i) {
#pragma unroll
    for (int j = 0; j < 2; ++j) {
      const size_t row = rowA0 + wave_m * 64 + i * 16 + orow;
      const size_t col = rowB0 + wave_n * 32 + j * 16 + ocol;
      float* op = Out + row * N + col;
#pragma unroll
      for (int e = 0; e < 8; ++e)
        op[(size_t)e * N] = acc[i][j][e] * INV_TEMP;
    }
  }
}

// ---------------------------------------------------------------------------
extern "C" void kernel_launch(void* const* d_in, const int* in_sizes, int n_in,
                              void* d_out, int out_size, void* d_ws, size_t ws_size,
                              hipStream_t stream) {
  const float* inputs   = (const float*)d_in[0];           // [B,D] fp32
  // d_in[1] = targets (unused by forward)
  const float* features = (const float*)d_in[2];           // [N,D] fp32

  const int B = in_sizes[1];                               // 1024
  const int D = in_sizes[0] / B;                           // 2048
  const int N = in_sizes[2] / D;                           // 16384

  const size_t needA = (size_t)B * D * sizeof(unsigned short);   // 4 MB
  const size_t needF = (size_t)N * D * sizeof(unsigned short);   // 64 MB

  unsigned short* xbf = (unsigned short*)d_ws;
  float* out = (float*)d_out;

  normalize_rows_bf16<<<B, 256, 0, stream>>>(inputs, xbf, D);

  dim3 grid(N / TILE_N, B / TILE_M);                       // (128, 8)

  if (ws_size >= needA + needF) {
    // Fast path: pre-convert the bank once; GEMM loop is loads+LDS+WMMA only.
    unsigned short* fbf = (unsigned short*)((char*)d_ws + needA);
    convert_f32_to_bf16<<<4096, 256, 0, stream>>>(features, fbf,
                                                  (size_t)N * D / 4);
    cosine_gemm_wmma<true><<<grid, 256, 0, stream>>>(xbf, fbf, nullptr,
                                                     out, D, N);
  } else {
    // Fallback: convert features fp32->bf16 while staging tiles into LDS.
    cosine_gemm_wmma<false><<<grid, 256, 0, stream>>>(xbf, nullptr, features,
                                                      out, D, N);
  }
}
